// ClusterCritic_85985245266042
// MI455X (gfx1250) — matrix operand order repaired and checked by reading, not compiled, not run on hardware
//
#include <hip/hip_runtime.h>

// ---- types for WMMA (gfx1250, wave32) ----
typedef __attribute__((ext_vector_type(4)))  _Float16 v4h;
typedef __attribute__((ext_vector_type(8)))  _Float16 v8h;
typedef __attribute__((ext_vector_type(16))) _Float16 v16h;
typedef __attribute__((ext_vector_type(8)))  float    v8f;
union V16 { v16h v; v8h h[2]; };

#define AGENTS 10
#define NCLUST 5
#define BATCH  32768
#define HID    32
#define NCH    16      // batch chunks for stats
#define TB     64      // batch tile of fused kernel

// ============================================================
// 1) per-(agent,feature) partial sum / sumsq, coalesced lanes
// ============================================================
__global__ __launch_bounds__(128) void stats_partial(
    const float* __restrict__ states, const float* __restrict__ actions,
    float* __restrict__ part)
{
  const int chunk = blockIdx.x, a = blockIdx.y, t = threadIdx.x;
  const int rows = BATCH / NCH;
  const int r0 = chunk * rows;
  float s = 0.f, sq = 0.f;
  if (t < 100) {
    const float* p = states + ((size_t)a * BATCH + r0) * 100 + t;
#pragma unroll 4
    for (int r = 0; r < rows; ++r) { float v = p[(size_t)r * 100]; s += v; sq += v * v; }
  } else if (t < 120) {
    const float* p = actions + ((size_t)a * BATCH + r0) * 20 + (t - 100);
#pragma unroll 4
    for (int r = 0; r < rows; ++r) { float v = p[(size_t)r * 20]; s += v; sq += v * v; }
  }
  size_t o = (((size_t)a * NCH + chunk) * 128 + t) * 2;
  part[o] = s; part[o + 1] = sq;
}

// ============================================================
// 2) deterministic reduce -> mean, rstd = rsqrt(var + eps)
// ============================================================
__global__ __launch_bounds__(128) void stats_finalize(
    const float* __restrict__ part, float* __restrict__ meanv, float* __restrict__ rstdv)
{
  const int a = blockIdx.x, t = threadIdx.x;
  float s = 0.f, sq = 0.f;
#pragma unroll
  for (int ch = 0; ch < NCH; ++ch) {
    size_t o = (((size_t)a * NCH + ch) * 128 + t) * 2;
    s += part[o]; sq += part[o + 1];
  }
  const float inv = 1.f / (float)BATCH;
  float m = s * inv;
  float var = sq * inv - m * m;
  meanv[a * 128 + t] = m;
  rstdv[a * 128 + t] = rsqrtf(var + 1e-5f);
}

// ============================================================
// 3) fold BN into encoder weights (f16, K padded to 256)
//    state-only encoder: zeros on action/pad columns
// ============================================================
__global__ __launch_bounds__(32) void fold_enc(
    const float* __restrict__ enc_W, const float* __restrict__ enc_b,
    const float* __restrict__ senc_W, const float* __restrict__ senc_b,
    const float* __restrict__ meanv, const float* __restrict__ rstdv,
    _Float16* __restrict__ wsa, _Float16* __restrict__ wss,
    float* __restrict__ bsa, float* __restrict__ bss)
{
  const int c = blockIdx.x, h = threadIdx.x;
  float accA = 0.f, accS = 0.f;
  for (int k = 0; k < 256; ++k) {
    float wa = 0.f, ws = 0.f;
    if (k < 240) {
      int ag, f;
      if (k < 100)      { ag = 2 * c;     f = k; }
      else if (k < 200) { ag = 2 * c + 1; f = k - 100; }
      else if (k < 220) { ag = 2 * c;     f = 100 + (k - 200); }
      else              { ag = 2 * c + 1; f = 100 + (k - 220); }
      float m  = meanv[ag * 128 + f];
      float rs = rstdv[ag * 128 + f];
      float we = enc_W[((size_t)c * HID + h) * 240 + k];
      wa = we * rs; accA += we * rs * m;
      if (k < 200) {
        float wv2 = senc_W[((size_t)c * HID + h) * 200 + k];
        ws = wv2 * rs; accS += wv2 * rs * m;
      }
    }
    wsa[((size_t)c * HID + h) * 256 + k] = (_Float16)wa;
    wss[((size_t)c * HID + h) * 256 + k] = (_Float16)ws;
  }
  bsa[c * HID + h] = enc_b[c * HID + h] - accA;
  bss[c * HID + h] = senc_b[c * HID + h] - accS;
}

// ============================================================
// 4) head weights -> f16 ([32 rows (n*8+d)][32 cols (hid)])
// ============================================================
__global__ __launch_bounds__(32) void fold_head(
    const float* __restrict__ key_W, const float* __restrict__ sel_W,
    const float* __restrict__ val_W, const float* __restrict__ val_b,
    _Float16* __restrict__ wk, _Float16* __restrict__ wv,
    _Float16* __restrict__ wsl, float* __restrict__ vbw)
{
  const int r = threadIdx.x;
#pragma unroll
  for (int k = 0; k < 32; ++k) {
    wk [r * 32 + k] = (_Float16)key_W[r * 32 + k];
    wsl[r * 32 + k] = (_Float16)sel_W[r * 32 + k];
    wv [r * 32 + k] = (_Float16)val_W[r * 32 + k];
  }
  vbw[r] = val_b[r];
}

// ============================================================
// 5) fused: gather+encode (WMMA) -> heads (WMMA) -> attention
//    256 threads = 8 wave32; dynamic LDS = 160 KB
// ============================================================
__device__ __forceinline__ v8f wmma_f16(const v16h& a, const v16h& b, const v8f& c) {
  return __builtin_amdgcn_wmma_f32_16x16x32_f16(false, a, false, b, (short)0, c, false, false);
}

__global__ __launch_bounds__(256) void fused_main(
    const float* __restrict__ states, const float* __restrict__ actions,
    const _Float16* __restrict__ wsa, const float* __restrict__ bsa,
    const _Float16* __restrict__ wss, const float* __restrict__ bss,
    const _Float16* __restrict__ wk, const _Float16* __restrict__ wv,
    const _Float16* __restrict__ wsl, const float* __restrict__ vbw,
    float* __restrict__ out)
{
  extern __shared__ char smem[];
  _Float16* xs   = (_Float16*)smem;             // [TB][256] gathered input, f16
  _Float16* esa  = xs  + TB * 256;              // [TB][32]  sa_enc, f16
  _Float16* ess  = esa + TB * 32;               // [TB][32]  s_enc,  f16
  float* keysL   = (float*)(ess + TB * 32);     // [5][TB][32]
  float* valsL   = keysL + NCLUST * TB * 32;
  float* selL    = valsL + NCLUST * TB * 32;

  const int tid  = threadIdx.x;
  const int lane = tid & 31, wave = tid >> 5;
  const int lh   = lane & 15, hb = lane >> 4;   // 16-lane half select
  const int b0   = blockIdx.x * TB;
  const int m0   = (wave & 3) * 16;             // batch-row tile
  const int n0   = (wave >> 2) * 16;            // output-col tile

  // zero pad columns 240..255 (never overwritten afterwards)
  if (tid < TB * 2) {
    int row = tid >> 1, seg = tid & 1;
    v8h z = {};
    *(v8h*)&xs[row * 256 + 240 + seg * 8] = z;
  }

  for (int c = 0; c < NCLUST; ++c) {
    __syncthreads();   // prior iteration's LDS reads complete
    // ---- gather + f32->f16 x tile: [row][agent-local features] ----
    for (int i = tid; i < TB * 50; i += 256) {
      int row = i / 50, q = i % 50, ag = q / 25, qq = q % 25;
      const float4 v = *reinterpret_cast<const float4*>(
          states + ((size_t)(2 * c + ag) * BATCH + b0 + row) * 100 + qq * 4);
      v4h hv; hv[0] = (_Float16)v.x; hv[1] = (_Float16)v.y;
              hv[2] = (_Float16)v.z; hv[3] = (_Float16)v.w;
      *(v4h*)&xs[row * 256 + ag * 100 + qq * 4] = hv;
    }
    for (int i = tid; i < TB * 10; i += 256) {
      int row = i / 10, q = i % 10, ag = q / 5, qq = q % 5;
      const float4 v = *reinterpret_cast<const float4*>(
          actions + ((size_t)(2 * c + ag) * BATCH + b0 + row) * 20 + qq * 4);
      v4h hv; hv[0] = (_Float16)v.x; hv[1] = (_Float16)v.y;
              hv[2] = (_Float16)v.z; hv[3] = (_Float16)v.w;
      *(v4h*)&xs[row * 256 + 200 + ag * 20 + qq * 4] = hv;
    }
    __syncthreads();

    // ---- encoder GEMMs: [TB,256]x[256,32] for sa_enc and s_enc ----
    v8f accA = {0.f,0.f,0.f,0.f,0.f,0.f,0.f,0.f};
    v8f accS = {0.f,0.f,0.f,0.f,0.f,0.f,0.f,0.f};
    const _Float16* xrow = xs + (m0 + lh) * 256;
    const _Float16* wra  = wsa + (size_t)(c * HID + n0 + lh) * 256;
    const _Float16* wrs  = wss + (size_t)(c * HID + n0 + lh) * 256;
#pragma unroll
    for (int k0 = 0; k0 < 256; k0 += 32) {
      V16 a, ba, bs2;
      a.h[0]   = *(const v8h*)(xrow + k0 + hb * 16);       // A: M per lane, contiguous K
      a.h[1]   = *(const v8h*)(xrow + k0 + hb * 16 + 8);
      ba.h[0]  = *(const v8h*)(wra + k0 + hb * 16);        // B: N per lane, contiguous K
      ba.h[1]  = *(const v8h*)(wra + k0 + hb * 16 + 8);
      bs2.h[0] = *(const v8h*)(wrs + k0 + hb * 16);
      bs2.h[1] = *(const v8h*)(wrs + k0 + hb * 16 + 8);
      accA = wmma_f16(a.v, ba.v, accA);
      accS = wmma_f16(a.v, bs2.v, accS);
    }
    {
      const float biasA = bsa[c * HID + n0 + lh];
      const float biasS = bss[c * HID + n0 + lh];
#pragma unroll
      for (int r = 0; r < 8; ++r) {
        int row = m0 + r + hb * 8;
        float va = accA[r] + biasA; va = va > 0.f ? va : 0.01f * va;
        float vs = accS[r] + biasS; vs = vs > 0.f ? vs : 0.01f * vs;
        esa[row * 32 + n0 + lh] = (_Float16)va;
        ess[row * 32 + n0 + lh] = (_Float16)vs;
      }
    }
    __syncthreads();

    // ---- head GEMMs (K=32, one WMMA each): keys, vals, sel ----
    V16 aa, as, bk, bv2, bsl;
    const _Float16* ea  = esa + (m0 + lh) * 32;
    const _Float16* es2 = ess + (m0 + lh) * 32;
    aa.h[0] = *(const v8h*)(ea + hb * 16);  aa.h[1] = *(const v8h*)(ea + hb * 16 + 8);
    as.h[0] = *(const v8h*)(es2 + hb * 16); as.h[1] = *(const v8h*)(es2 + hb * 16 + 8);
    const _Float16* pk = wk  + (n0 + lh) * 32 + hb * 16;
    const _Float16* pv = wv  + (n0 + lh) * 32 + hb * 16;
    const _Float16* ps = wsl + (n0 + lh) * 32 + hb * 16;
    bk.h[0]  = *(const v8h*)pk; bk.h[1]  = *(const v8h*)(pk + 8);
    bv2.h[0] = *(const v8h*)pv; bv2.h[1] = *(const v8h*)(pv + 8);
    bsl.h[0] = *(const v8h*)ps; bsl.h[1] = *(const v8h*)(ps + 8);
    v8f zr = {0.f,0.f,0.f,0.f,0.f,0.f,0.f,0.f};
    v8f ck = wmma_f16(aa.v, bk.v,  zr);
    v8f cv = wmma_f16(aa.v, bv2.v, zr);
    v8f cs = wmma_f16(as.v, bsl.v, zr);
    const float vbias = vbw[n0 + lh];
#pragma unroll
    for (int r = 0; r < 8; ++r) {
      int row = m0 + r + hb * 8, col = n0 + lh;
      keysL[(c * TB + row) * 32 + col] = ck[r];
      float t = cv[r] + vbias;
      valsL[(c * TB + row) * 32 + col] = t > 0.f ? t : 0.01f * t;
      selL [(c * TB + row) * 32 + col] = cs[r];
    }
  }
  __syncthreads();

  // ---- attention: one thread per (batch element, head) ----
  {
    const int bl = tid & 63;
    const int n  = tid >> 6;               // head 0..3
    const size_t bg = (size_t)b0 + bl;
    float key[5][8], val[5][8], sl[5][8];
#pragma unroll
    for (int j = 0; j < 5; ++j) {
      const float* kp = &keysL[(j * TB + bl) * 32 + n * 8];
      const float* vp = &valsL[(j * TB + bl) * 32 + n * 8];
      const float* sp = &selL [(j * TB + bl) * 32 + n * 8];
      float4 a0 = *(const float4*)kp, a1 = *(const float4*)(kp + 4);
      key[j][0]=a0.x; key[j][1]=a0.y; key[j][2]=a0.z; key[j][3]=a0.w;
      key[j][4]=a1.x; key[j][5]=a1.y; key[j][6]=a1.z; key[j][7]=a1.w;
      float4 b0v = *(const float4*)vp, b1v = *(const float4*)(vp + 4);
      val[j][0]=b0v.x; val[j][1]=b0v.y; val[j][2]=b0v.z; val[j][3]=b0v.w;
      val[j][4]=b1v.x; val[j][5]=b1v.y; val[j][6]=b1v.z; val[j][7]=b1v.w;
      float4 c0 = *(const float4*)sp, c1 = *(const float4*)(sp + 4);
      sl[j][0]=c0.x; sl[j][1]=c0.y; sl[j][2]=c0.z; sl[j][3]=c0.w;
      sl[j][4]=c1.x; sl[j][5]=c1.y; sl[j][6]=c1.z; sl[j][7]=c1.w;
    }
    float* out_attn = out;
    float* out_lg   = out + (size_t)NCLUST * 4 * BATCH * 8;
    float* out_pr   = out_lg + (size_t)NCLUST * 4 * BATCH * 4;
    const float scale = 0.35355339059327373f;  // 1/sqrt(8)
#pragma unroll
    for (int i = 0; i < 5; ++i) {
      float lg[5];
#pragma unroll
      for (int j = 0; j < 5; ++j) {
        float d = 0.f;
#pragma unroll
        for (int u = 0; u < 8; ++u) d += sl[i][u] * key[j][u];
        lg[j] = d;
      }
      float mx = -3.4e38f;
#pragma unroll
      for (int j = 0; j < 5; ++j) if (j != i) { float v = lg[j] * scale; mx = v > mx ? v : mx; }
      float e[5], ssum = 0.f;
#pragma unroll
      for (int j = 0; j < 5; ++j) { e[j] = (j == i) ? 0.f : __expf(lg[j] * scale - mx); ssum += e[j]; }
      const float inv = 1.f / ssum;
      float at[8] = {0.f,0.f,0.f,0.f,0.f,0.f,0.f,0.f};
#pragma unroll
      for (int j = 0; j < 5; ++j) if (j != i) {
        float p = e[j] * inv;
#pragma unroll
        for (int d = 0; d < 8; ++d) at[d] += p * val[j][d];
      }
      const size_t base = (size_t)(i * 4 + n) * BATCH + bg;
      *(float4*)&out_attn[base * 8]     = make_float4(at[0], at[1], at[2], at[3]);
      *(float4*)&out_attn[base * 8 + 4] = make_float4(at[4], at[5], at[6], at[7]);
      float lo[4], po[4]; int idx = 0;
#pragma unroll
      for (int j = 0; j < 5; ++j) if (j != i) { lo[idx] = lg[j]; po[idx] = e[j] * inv; ++idx; }
      *(float4*)&out_lg[base * 4] = make_float4(lo[0], lo[1], lo[2], lo[3]);
      *(float4*)&out_pr[base * 4] = make_float4(po[0], po[1], po[2], po[3]);
    }
  }
}

// ============================================================
extern "C" void kernel_launch(void* const* d_in, const int* in_sizes, int n_in,
                              void* d_out, int out_size, void* d_ws, size_t ws_size,
                              hipStream_t stream) {
  (void)in_sizes; (void)n_in; (void)out_size; (void)ws_size;
  const float* states = (const float*)d_in[0];
  const float* actions = (const float*)d_in[1];
  const float* enc_W  = (const float*)d_in[2];
  const float* enc_b  = (const float*)d_in[3];
  const float* senc_W = (const float*)d_in[4];
  const float* senc_b = (const float*)d_in[5];
  const float* key_W  = (const float*)d_in[6];
  const float* sel_W  = (const float*)d_in[7];
  const float* val_W  = (const float*)d_in[8];
  const float* val_b  = (const float*)d_in[9];
  float* out = (float*)d_out;

  // ---- workspace carve (fully rewritten every call; no atomics) ----
  float* part  = (float*)d_ws;                       // 10*16*128*2 = 40960 f
  float* meanv = part + (size_t)AGENTS * NCH * 128 * 2;  // 10*128
  float* rstdv = meanv + AGENTS * 128;               // 10*128
  float* bsa   = rstdv + AGENTS * 128;               // 160
  float* bss   = bsa + NCLUST * HID;                 // 160
  float* vbw   = bss + NCLUST * HID;                 // 32
  _Float16* wsa = (_Float16*)(vbw + 32);             // 5*32*256 halfs
  _Float16* wss = wsa + (size_t)NCLUST * HID * 256;
  _Float16* wk  = wss + (size_t)NCLUST * HID * 256;  // 32*32
  _Float16* wv  = wk + 1024;
  _Float16* wsl = wv + 1024;

  stats_partial<<<dim3(NCH, AGENTS), 128, 0, stream>>>(states, actions, part);
  stats_finalize<<<AGENTS, 128, 0, stream>>>(part, meanv, rstdv);
  fold_enc<<<NCLUST, 32, 0, stream>>>(enc_W, enc_b, senc_W, senc_b, meanv, rstdv,
                                      wsa, wss, bsa, bss);
  fold_head<<<1, 32, 0, stream>>>(key_W, sel_W, val_W, val_b, wk, wv, wsl, vbw);

  const size_t smem = (size_t)TB * 256 * 2      // x tile f16
                    + 2 * (size_t)TB * 32 * 2   // sa_enc + s_enc f16
                    + 3 * (size_t)NCLUST * TB * 32 * 4;  // keys/vals/sel f32 = 160 KB
  fused_main<<<BATCH / TB, 256, smem, stream>>>(states, actions, wsa, bsa, wss, bss,
                                                wk, wv, wsl, vbw, out);
}
